// _CausalTimeAttention_3693671875185
// MI455X (gfx1250) — compile-verified, matrix operand order checked by reading, not run on hardware
//
#include <hip/hip_runtime.h>

// ---------------------------------------------------------------------------
// CausalTimeAttention for MI455X (gfx1250, wave32, WMMA 16x16x32 bf16)
//
// Shapes: b=2, c=512, f=64, t=512, heads=8, hd=64, window=128.
//   N  = b*f = 128 sequences, T = 512, C = 512, M1 = N*T = 65536.
// All GEMMs use v_wmma_f32_16x16x32_bf16 (fp32 accumulate), softmax in fp32.
//
// Fragment layout assumptions (CDNA5 ISA 7.12.2, 16-bit A 16x32):
//   lanes 0-15 : row = lane,    VGPR0-3 = K 0-7,  VGPR4-7 = K 16-23
//   lanes 16-31: row = lane-16, VGPR0-3 = K 8-15, VGPR4-7 = K 24-31
// B (32x16) assumed mirror layout with lane -> column N, VGPR pairs -> K.
// Hence both A and B fragments load as two contiguous 16B chunks per lane
// when the K dimension is contiguous in memory (we lay out all operands so).
// ---------------------------------------------------------------------------

typedef __attribute__((ext_vector_type(16))) __bf16 v16bf;
typedef __attribute__((ext_vector_type(8)))  float  v8f;

union Frag16 { uint4 q[2]; v16bf v; };

#define NEG_BIG (-1.0e30f)   // finite "-inf": expf underflows to 0, no NaN path

__device__ __forceinline__ unsigned short f32_to_bf16(float f) {
  union { float f; unsigned u; } a; a.f = f;
  unsigned r = a.u + 0x7FFFu + ((a.u >> 16) & 1u);   // round-to-nearest-even
  return (unsigned short)(r >> 16);
}

__device__ __forceinline__ v8f wmma_bf16(v16bf a, v16bf b, v8f c) {
  return __builtin_amdgcn_wmma_f32_16x16x32_bf16(
      /*neg_a=*/false, a, /*neg_b=*/false, b,
      /*c_mod=*/(short)0, c, /*reuse_a=*/false, /*reuse_b=*/false);
}

// Load a 16(x-rows) x 32(K) bf16 fragment whose K dimension is contiguous.
// base points at (row 0, K 0) of the tile; stride is in bf16 elements.
__device__ __forceinline__ v16bf load_frag(const unsigned short* base,
                                           int stride, int lane) {
  int r  = lane & 15;
  int kb = (lane >> 4) << 3;           // 0 or 8
  const unsigned short* p = base + r * stride;
  Frag16 f;
  f.q[0] = *(const uint4*)(p + kb);        // K kb..kb+7
  f.q[1] = *(const uint4*)(p + kb + 16);   // K 16+kb..16+kb+7
  return f.v;
}

// ---------------------------------------------------------------------------
// Kernel 1: x (b,c,f,t) f32  ->  Hb (n=b*f, t, c) bf16
// ---------------------------------------------------------------------------
__global__ __launch_bounds__(256) void cta_convert_x(
    const float* __restrict__ x, unsigned short* __restrict__ Hb) {
  unsigned idx = blockIdx.x * 256u + threadIdx.x;    // 33,554,432 total
  int t = idx & 511;
  int f = (idx >> 9) & 63;
  int c = (idx >> 15) & 511;
  int b = idx >> 24;
  Hb[(((unsigned)(b * 64 + f) * 512u + t) * 512u) + c] = f32_to_bf16(x[idx]);
}

// ---------------------------------------------------------------------------
// Kernel 2: W (K,N) f32 -> WT (N,K) bf16   (K-contiguous for B fragments)
// ---------------------------------------------------------------------------
__global__ __launch_bounds__(256) void cta_convert_w(
    const float* __restrict__ W, unsigned short* __restrict__ WT,
    int K, int Ncols) {
  int idx = blockIdx.x * 256 + threadIdx.x;
  if (idx >= K * Ncols) return;
  int n = idx % Ncols;
  int k = idx / Ncols;
  WT[n * K + k] = f32_to_bf16(W[idx]);
}

// ---------------------------------------------------------------------------
// Kernel 3: QKV GEMM.  Hb(65536x512) @ WqkvT'(512x1536) + bias.
// Wave computes a 32x32 tile. Stores Q,K as (n,h,t,d); V transposed (n,h,d,t).
// ---------------------------------------------------------------------------
__global__ __launch_bounds__(256) void cta_gemm_qkv(
    const unsigned short* __restrict__ Hb,
    const unsigned short* __restrict__ WT,     // (1536, 512) bf16
    const float* __restrict__ bias,            // (1536,)
    unsigned short* __restrict__ Qb,
    unsigned short* __restrict__ Kb,
    unsigned short* __restrict__ VT) {
  int lane = threadIdx.x & 31;
  int wave = threadIdx.x >> 5;
  int n0 = (blockIdx.x * 8 + wave) * 32;   // [0, 1536)
  int m0 = blockIdx.y * 32;                // [0, 65536)

  v8f acc[2][2] = {};
#pragma unroll 4
  for (int k0 = 0; k0 < 512; k0 += 32) {
    v16bf a0 = load_frag(Hb + (unsigned)(m0 +  0) * 512u + k0, 512, lane);
    v16bf a1 = load_frag(Hb + (unsigned)(m0 + 16) * 512u + k0, 512, lane);
    v16bf b0 = load_frag(WT + (unsigned)(n0 +  0) * 512u + k0, 512, lane);
    v16bf b1 = load_frag(WT + (unsigned)(n0 + 16) * 512u + k0, 512, lane);
    acc[0][0] = wmma_bf16(a0, b0, acc[0][0]);
    acc[0][1] = wmma_bf16(a0, b1, acc[0][1]);
    acc[1][0] = wmma_bf16(a1, b0, acc[1][0]);
    acc[1][1] = wmma_bf16(a1, b1, acc[1][1]);
  }

  int lcol    = lane & 15;
  int rowbase = m0 + ((lane >> 4) << 3);
#pragma unroll
  for (int mi = 0; mi < 2; ++mi) {
#pragma unroll
    for (int ni = 0; ni < 2; ++ni) {
      int col    = n0 + ni * 16 + lcol;         // [0,1536)
      float bv   = bias[col];
      int mat    = col >> 9;                    // 0=Q 1=K 2=V
      int within = col & 511;
      int head   = within >> 6;
      int d      = within & 63;
#pragma unroll
      for (int r = 0; r < 8; ++r) {
        int m    = rowbase + mi * 16 + r;
        int nseq = m >> 9;
        int t    = m & 511;
        unsigned short bf = f32_to_bf16(acc[mi][ni][r] + bv);
        if (mat == 0)
          Qb[(((unsigned)(nseq * 8 + head) * 512u + t) * 64u) + d] = bf;
        else if (mat == 1)
          Kb[(((unsigned)(nseq * 8 + head) * 512u + t) * 64u) + d] = bf;
        else
          VT[(((unsigned)(nseq * 8 + head) * 64u + d) * 512u) + t] = bf;
      }
    }
  }
}

// ---------------------------------------------------------------------------
// Kernel 4: banded flash attention. One wave per (nseq, head, 16-query tile).
// Keys restricted to the causal window [q-127, q]; processed 32 at a time.
// All 8 waves of a block share (nseq, head): rel bias staged in LDS once.
// ---------------------------------------------------------------------------
__global__ __launch_bounds__(256) void cta_attn(
    const unsigned short* __restrict__ Qb,
    const unsigned short* __restrict__ Kb,
    const unsigned short* __restrict__ VT,
    const float* __restrict__ rel,             // (8, 128)
    unsigned short* __restrict__ Yb) {         // (n, t, c) bf16
  __shared__ __align__(16) unsigned short pbuf[8][16][40];  // 80B row stride
  __shared__ float relbuf[128];

  int lane = threadIdx.x & 31;
  int wave = threadIdx.x >> 5;
  int wid  = blockIdx.x * 8 + wave;            // 0..32767
  int qt   = wid & 31;
  int head = (wid >> 5) & 7;                   // constant across the block
  int nseq = wid >> 8;                         // constant across the block
  int q0   = qt * 16;
  int half = lane >> 4;
  int lcol = lane & 15;

  if (threadIdx.x < 128) relbuf[threadIdx.x] = rel[head * 128 + threadIdx.x];
  __syncthreads();

  const unsigned short* Q2 = Qb + (unsigned)(nseq * 8 + head) * 512u * 64u;
  const unsigned short* K2 = Kb + (unsigned)(nseq * 8 + head) * 512u * 64u;
  const unsigned short* V2 = VT + (unsigned)(nseq * 8 + head) * 64u * 512u;

  v16bf aq0 = load_frag(Q2 + (unsigned)q0 * 64u +  0, 64, lane);
  v16bf aq1 = load_frag(Q2 + (unsigned)q0 * 64u + 32, 64, lane);

  v8f o[4] = {};
  float rowm[8], rowl[8];
#pragma unroll
  for (int r = 0; r < 8; ++r) { rowm[r] = NEG_BIG; rowl[r] = 0.f; }

  int kstart = q0 - 127;
  if (kstart < 0) kstart = 0;
  kstart &= ~31;

  for (int k0 = kstart; k0 <= q0 + 15; k0 += 32) {
    // ---- scores: S = Q (16x64) @ K^T (64x32), two 16-key groups ----
    v8f s[2];
#pragma unroll
    for (int g = 0; g < 2; ++g) {
      const unsigned short* kb = K2 + (unsigned)(k0 + g * 16) * 64u;
      v16bf b0 = load_frag(kb +  0, 64, lane);
      v16bf b1 = load_frag(kb + 32, 64, lane);
      v8f z = {};
      s[g] = wmma_bf16(aq1, b1, wmma_bf16(aq0, b0, z));
    }

    // ---- bias + causal-band mask + online softmax stats ----
#pragma unroll
    for (int r = 0; r < 8; ++r) {
      int q = q0 + r + 8 * half;
      int key0 = k0 + lcol;
      int key1 = key0 + 16;
      int d0 = q - key0, d1 = q - key1;
      // Masked entries get a finite very-negative sentinel: expf underflows
      // to 0; rows that were fully masked so far are wiped by alpha==0 the
      // moment their first valid tile arrives (key q is always valid).
      float e0 = ((unsigned)d0 < 128u) ? (s[0][r] * 0.125f + relbuf[d0]) : NEG_BIG;
      float e1 = ((unsigned)d1 < 128u) ? (s[1][r] * 0.125f + relbuf[d1]) : NEG_BIG;
      float mr = fmaxf(e0, e1);
      mr = fmaxf(mr, __shfl_xor(mr, 1, 32));
      mr = fmaxf(mr, __shfl_xor(mr, 2, 32));
      mr = fmaxf(mr, __shfl_xor(mr, 4, 32));
      mr = fmaxf(mr, __shfl_xor(mr, 8, 32));
      float mnew  = fmaxf(rowm[r], mr);
      float alpha = __expf(rowm[r] - mnew);   // 1 if both NEG_BIG, 0 on revive
      float p0 = __expf(e0 - mnew);
      float p1 = __expf(e1 - mnew);
      float rs = p0 + p1;
      rs += __shfl_xor(rs, 1, 32);
      rs += __shfl_xor(rs, 2, 32);
      rs += __shfl_xor(rs, 4, 32);
      rs += __shfl_xor(rs, 8, 32);
      rowl[r] = rowl[r] * alpha + rs;
      rowm[r] = mnew;
#pragma unroll
      for (int dt = 0; dt < 4; ++dt) o[dt][r] *= alpha;
      int prow = r + 8 * half;
      pbuf[wave][prow][lcol]      = f32_to_bf16(p0);
      pbuf[wave][prow][lcol + 16] = f32_to_bf16(p1);
    }

    // ---- transpose P (C layout -> A fragment) via wave-private LDS ----
    Frag16 pf;
    {
      int r2  = lane & 15;
      int kb2 = (lane >> 4) << 3;
      pf.q[0] = *(const uint4*)&pbuf[wave][r2][kb2];
      pf.q[1] = *(const uint4*)&pbuf[wave][r2][kb2 + 16];
    }

    // ---- O += P (16x32) @ V (32x64) ----
#pragma unroll
    for (int dt = 0; dt < 4; ++dt) {
      v16bf bv = load_frag(V2 + (unsigned)(dt * 16) * 512u + k0, 512, lane);
      o[dt] = wmma_bf16(pf.v, bv, o[dt]);
    }
  }

  // ---- normalize and store Y (n, t, c=h*64+d) bf16 ----
  float invl[8];
#pragma unroll
  for (int r = 0; r < 8; ++r) invl[r] = 1.0f / rowl[r];
#pragma unroll
  for (int dt = 0; dt < 4; ++dt) {
    int col = head * 64 + dt * 16 + lcol;
#pragma unroll
    for (int r = 0; r < 8; ++r) {
      int q = q0 + r + 8 * half;
      Yb[((unsigned)(nseq * 512 + q) * 512u) + col] = f32_to_bf16(o[dt][r] * invl[r]);
    }
  }
}

// ---------------------------------------------------------------------------
// Kernel 5: output projection. Yb(65536x512) @ WprojT'(512x512) + bias,
// scatter-stored as out(b, c, f, t) fp32.
// ---------------------------------------------------------------------------
__global__ __launch_bounds__(256) void cta_gemm_proj(
    const unsigned short* __restrict__ Yb,
    const unsigned short* __restrict__ WT,     // (512, 512) bf16
    const float* __restrict__ bias,            // (512,)
    float* __restrict__ out) {
  int lane = threadIdx.x & 31;
  int wave = threadIdx.x >> 5;
  int n0 = (blockIdx.x * 8 + wave) * 32;   // [0, 512)
  int m0 = blockIdx.y * 32;                // [0, 65536)

  v8f acc[2][2] = {};
#pragma unroll 4
  for (int k0 = 0; k0 < 512; k0 += 32) {
    v16bf a0 = load_frag(Yb + (unsigned)(m0 +  0) * 512u + k0, 512, lane);
    v16bf a1 = load_frag(Yb + (unsigned)(m0 + 16) * 512u + k0, 512, lane);
    v16bf b0 = load_frag(WT + (unsigned)(n0 +  0) * 512u + k0, 512, lane);
    v16bf b1 = load_frag(WT + (unsigned)(n0 + 16) * 512u + k0, 512, lane);
    acc[0][0] = wmma_bf16(a0, b0, acc[0][0]);
    acc[0][1] = wmma_bf16(a0, b1, acc[0][1]);
    acc[1][0] = wmma_bf16(a1, b0, acc[1][0]);
    acc[1][1] = wmma_bf16(a1, b1, acc[1][1]);
  }

  int lcol    = lane & 15;
  int rowbase = m0 + ((lane >> 4) << 3);
#pragma unroll
  for (int mi = 0; mi < 2; ++mi) {
#pragma unroll
    for (int ni = 0; ni < 2; ++ni) {
      int col  = n0 + ni * 16 + lcol;           // output channel c
      float bv = bias[col];
#pragma unroll
      for (int r = 0; r < 8; ++r) {
        int m    = rowbase + mi * 16 + r;
        int nseq = m >> 9;
        int t    = m & 511;
        int b    = nseq >> 6;
        int f    = nseq & 63;
        out[(((unsigned)(b * 512 + col) * 64u + f) * 512u) + t] =
            acc[mi][ni][r] + bv;
      }
    }
  }
}

// ---------------------------------------------------------------------------
extern "C" void kernel_launch(void* const* d_in, const int* in_sizes, int n_in,
                              void* d_out, int out_size, void* d_ws, size_t ws_size,
                              hipStream_t stream) {
  const float* x      = (const float*)d_in[0];
  const float* W_qkv  = (const float*)d_in[1];
  const float* b_qkv  = (const float*)d_in[2];
  const float* W_proj = (const float*)d_in[3];
  const float* b_proj = (const float*)d_in[4];
  const float* rel    = (const float*)d_in[5];
  float* out          = (float*)d_out;

  // Workspace layout (bytes). Yb reuses the Hb region (Hb dead after GEMM1).
  char* ws = (char*)d_ws;
  const size_t szHb = 65536ull * 512ull * 2ull;          // 64 MiB (also Yb)
  const size_t szWq = 1536ull * 512ull * 2ull;           // 1.5 MiB
  const size_t szWp = 512ull * 512ull * 2ull;            // 0.5 MiB
  const size_t szQ  = 128ull * 8ull * 512ull * 64ull * 2ull; // 64 MiB each
  unsigned short* Hb     = (unsigned short*)(ws);
  unsigned short* Yb     = Hb;                           // alias (stream-ordered)
  unsigned short* WqkvT  = (unsigned short*)(ws + szHb);
  unsigned short* WprojT = (unsigned short*)(ws + szHb + szWq);
  unsigned short* Qb     = (unsigned short*)(ws + szHb + szWq + szWp);
  unsigned short* Kb     = (unsigned short*)(ws + szHb + szWq + szWp + szQ);
  unsigned short* VT     = (unsigned short*)(ws + szHb + szWq + szWp + 2 * szQ);

  // 1) x -> Hb bf16 (n,t,c)
  cta_convert_x<<<33554432 / 256, 256, 0, stream>>>(x, Hb);
  // 2) weights -> transposed bf16
  cta_convert_w<<<(512 * 1536 + 255) / 256, 256, 0, stream>>>(W_qkv, WqkvT, 512, 1536);
  cta_convert_w<<<(512 * 512 + 255) / 256, 256, 0, stream>>>(W_proj, WprojT, 512, 512);
  // 3) QKV GEMM (grid: 48 n-tiles of 32 / 8 waves, 2048 m-tiles of 32)
  cta_gemm_qkv<<<dim3(6, 2048), 256, 0, stream>>>(Hb, WqkvT, b_qkv, Qb, Kb, VT);
  // 4) banded flash attention (32768 wave-tiles / 8 waves per block)
  cta_attn<<<4096, 256, 0, stream>>>(Qb, Kb, VT, rel, Yb);
  // 5) projection GEMM (16 n-tiles of 32 / 8 waves, 2048 m-tiles)
  cta_gemm_proj<<<dim3(2, 2048), 256, 0, stream>>>(Yb, WprojT, b_proj, out);
}
